// NanoGPTBlock_54863912239619
// MI455X (gfx1250) — matrix operand order repaired and checked by reading, not compiled, not run on hardware
//
#include <hip/hip_runtime.h>

// ---------------------------------------------------------------------------
// NanoGPT block forward on MI455X (gfx1250, wave32, WMMA).
// fp32 inputs; activations/weights downconverted to f16 for the matrix pipes,
// fp32 accumulation via v_wmma_f32_16x16x32_f16. Tile staging uses the CDNA5
// async global->LDS path (ASYNCcnt) with double-buffered LDS tiles.
// ---------------------------------------------------------------------------

typedef _Float16 half_t;
typedef __attribute__((ext_vector_type(16))) _Float16 v16h;
typedef __attribute__((ext_vector_type(8)))  _Float16 v8h;
typedef __attribute__((ext_vector_type(8)))  float    v8f;

static constexpr int Bx = 4;     // batch
static constexpr int Tx = 2048;  // sequence
static constexpr int Dx = 1024;  // model dim

static constexpr int BM = 128;   // block tile M
static constexpr int BN = 128;   // block tile N
static constexpr int KT = 32;    // K step (WMMA f16 K)

// ---------------------------------------------------------------------------
// CDNA5 async copy: global -> LDS, 16 bytes per lane, tracked by ASYNCcnt.
// Generic pointers to __shared__ carry the LDS byte offset in their low 32
// bits (ISA 10.2).
// ---------------------------------------------------------------------------
__device__ __forceinline__ void async_copy16(void* lds_dst, const half_t* src)
{
    uint32_t dst = (uint32_t)(uintptr_t)lds_dst;
    asm volatile("global_load_async_to_lds_b128 %0, %1, off"
                 :: "v"(dst), "v"(src));
}
__device__ __forceinline__ void wait_async0()
{
    asm volatile("s_wait_asynccnt 0x0" ::: "memory");
}

// ---------------------------------------------------------------------------
// Weight prep: fp32 [K][N] row-major -> f16 [N][K] row-major (transposed),
// so every GEMM consumes B in the fast NT layout.
// ---------------------------------------------------------------------------
__global__ __launch_bounds__(256)
void transpose_f32_to_f16(const float* __restrict__ in, half_t* __restrict__ out,
                          int K, int N)
{
    __shared__ float tile[32][33];
    const int k0 = blockIdx.y * 32;
    const int n0 = blockIdx.x * 32;
    const int tx = threadIdx.x & 31;
    const int ty = threadIdx.x >> 5;           // 8 row-groups
    #pragma unroll
    for (int r = ty; r < 32; r += 8)
        tile[r][tx] = in[(long)(k0 + r) * N + n0 + tx];
    __syncthreads();
    #pragma unroll
    for (int r = ty; r < 32; r += 8)
        out[(long)(n0 + r) * K + k0 + tx] = (half_t)tile[tx][r];
}

// ---------------------------------------------------------------------------
// LayerNorm (faithful to reference's buggy form):
//   out = (x - mean/sqrt(var_unbiased)) * gamma + beta   -> f16
// One block (256 threads) per row of D=1024.
// ---------------------------------------------------------------------------
__global__ __launch_bounds__(256)
void ln_f16_kernel(const float* __restrict__ x, const float* __restrict__ gamma,
                   const float* __restrict__ beta, half_t* __restrict__ out, int D)
{
    const long row = blockIdx.x;
    const float* xr = x + row * D;
    half_t* orow = out + row * D;
    __shared__ float red[256];

    float s = 0.f, s2 = 0.f;
    for (int c = threadIdx.x; c < D; c += 256) {
        float v = xr[c];
        s += v; s2 += v * v;
    }
    red[threadIdx.x] = s; __syncthreads();
    for (int o = 128; o > 0; o >>= 1) {
        if (threadIdx.x < o) red[threadIdx.x] += red[threadIdx.x + o];
        __syncthreads();
    }
    const float mean = red[0] / (float)D;
    __syncthreads();
    red[threadIdx.x] = s2; __syncthreads();
    for (int o = 128; o > 0; o >>= 1) {
        if (threadIdx.x < o) red[threadIdx.x] += red[threadIdx.x + o];
        __syncthreads();
    }
    const float sumsq = red[0];
    const float var = (sumsq - (float)D * mean * mean) / (float)(D - 1);
    const float sub = mean * rsqrtf(var);   // mu / sqrt(var)

    for (int c = threadIdx.x; c < D; c += 256)
        orow[c] = (half_t)((xr[c] - sub) * gamma[c] + beta[c]);
}

// ---------------------------------------------------------------------------
// Row softmax: f32 scores (causal-masked with -1e30) -> f16 probs.
// One block per row of width W=2048.
// ---------------------------------------------------------------------------
__global__ __launch_bounds__(256)
void softmax_f16_kernel(const float* __restrict__ S, half_t* __restrict__ P, int W)
{
    const long row = blockIdx.x;
    const float* sr = S + row * W;
    half_t* pr = P + row * W;
    __shared__ float red[256];

    float m = -3.4e38f;
    for (int c = threadIdx.x; c < W; c += 256) m = fmaxf(m, sr[c]);
    red[threadIdx.x] = m; __syncthreads();
    for (int o = 128; o > 0; o >>= 1) {
        if (threadIdx.x < o) red[threadIdx.x] = fmaxf(red[threadIdx.x], red[threadIdx.x + o]);
        __syncthreads();
    }
    m = red[0]; __syncthreads();

    float sum = 0.f;
    for (int c = threadIdx.x; c < W; c += 256) sum += __expf(sr[c] - m);
    red[threadIdx.x] = sum; __syncthreads();
    for (int o = 128; o > 0; o >>= 1) {
        if (threadIdx.x < o) red[threadIdx.x] += red[threadIdx.x + o];
        __syncthreads();
    }
    const float inv = 1.f / red[0];

    for (int c = threadIdx.x; c < W; c += 256)
        pr[c] = (half_t)(__expf(sr[c] - m) * inv);
}

// ---------------------------------------------------------------------------
// Tiled WMMA GEMM:  C[M,N] = A[M,K] * B^T   (B stored [N,K] row-major; lda/ldb
// are the row strides of A / B in elements).
// Block: 256 threads = 8 waves; tile 128x128; each wave owns 32x64
// (2 x 4 fragments of 16x16), K-loop step 32 using v_wmma_f32_16x16x32_f16.
// Global->LDS staging via global_load_async_to_lds_b128 with double-buffered
// LDS tiles (compute on buf, async-fill buf^1, s_wait_asynccnt + barrier).
//
// Epilogues (EPI):
//   0: f16 store C[row*N+col]                      (Q,K projections)
//   1: f32 store, *scale, causal mask col>row      (attention scores)
//   2: f32 store, + extra[idx]                     (PV + x residual -> y)
//   3: f16 store, + bias[col], ReLU                (MLP layer 1)
//   4: f32 store, + bias[col] + extra[idx]         (MLP layer 2 + y -> out)
//   5: f16 transposed store C[col*M+row]           (V projection -> V^T)
// blockIdx.z batches with explicit element strides.
// ---------------------------------------------------------------------------
template<int EPI>
__global__ __launch_bounds__(256)
void gemm_wmma(const half_t* __restrict__ Ag, const half_t* __restrict__ Bg,
               float* __restrict__ Cf, half_t* __restrict__ Ch,
               const float* __restrict__ bias, const float* __restrict__ extra,
               float scale, int M, int N, int K, int lda, int ldb,
               long batchA, long batchB, long batchC)
{
    Ag += (long)blockIdx.z * batchA;
    Bg += (long)blockIdx.z * batchB;
    if (EPI == 1 || EPI == 2 || EPI == 4)  Cf += (long)blockIdx.z * batchC;
    if (EPI == 0 || EPI == 3 || EPI == 5)  Ch += (long)blockIdx.z * batchC;
    if (EPI == 2 || EPI == 4)              extra += (long)blockIdx.z * batchC;

    __shared__ __align__(128) half_t As[2][BM][KT];
    __shared__ __align__(128) half_t Bs[2][BN][KT];

    const int tid  = threadIdx.x;
    const int lane = tid & 31;
    const int wave = tid >> 5;
    const int wm   = wave & 3;        // 4 groups of 32 rows
    const int wn   = wave >> 2;       // 2 groups of 64 cols
    const int lh   = lane & 15;       // lane within half-wave
    const int hi   = lane >> 4;       // 0 / 1

    const int rowBase = blockIdx.x * BM;
    const int colBase = blockIdx.y * BN;

    // per-thread staging addresses: thread -> rows rS and rS+64, 16B column kc
    const int rS = tid >> 2;              // 0..63
    const int kc = (tid & 3) * 8;         // 0,8,16,24
    const half_t* aSrc0 = Ag + (long)(rowBase + rS) * lda + kc;
    const half_t* aSrc1 = aSrc0 + (long)64 * lda;
    const half_t* bSrc0 = Bg + (long)(colBase + rS) * ldb + kc;
    const half_t* bSrc1 = bSrc0 + (long)64 * ldb;

    v8f acc[2][4];
    #pragma unroll
    for (int mt = 0; mt < 2; ++mt)
        #pragma unroll
        for (int nt = 0; nt < 4; ++nt)
            acc[mt][nt] = (v8f){0.f,0.f,0.f,0.f,0.f,0.f,0.f,0.f};

    const int kTiles = K / KT;

    // prologue: fill buffer 0
    async_copy16(&As[0][rS     ][kc], aSrc0);
    async_copy16(&As[0][rS + 64][kc], aSrc1);
    async_copy16(&Bs[0][rS     ][kc], bSrc0);
    async_copy16(&Bs[0][rS + 64][kc], bSrc1);
    wait_async0();
    __syncthreads();

    int buf = 0;
    for (int kt = 0; kt < kTiles; ++kt) {
        // kick off next tile into the other buffer while we compute
        if (kt + 1 < kTiles) {
            const int k1 = (kt + 1) * KT;
            async_copy16(&As[buf ^ 1][rS     ][kc], aSrc0 + k1);
            async_copy16(&As[buf ^ 1][rS + 64][kc], aSrc1 + k1);
            async_copy16(&Bs[buf ^ 1][rS     ][kc], bSrc0 + k1);
            async_copy16(&Bs[buf ^ 1][rS + 64][kc], bSrc1 + k1);
        }

        // B fragments: 32x16, one contiguous 32B LDS read each
        v16h bfv[4];
        #pragma unroll
        for (int nt = 0; nt < 4; ++nt)
            bfv[nt] = *(const v16h*)&Bs[buf][wn * 64 + nt * 16 + lh][hi * 16];

        // A fragments (contiguous k order, same as B: lanes 0-15 row M=lane, K=0..15;
        // lanes 16-31 same rows, K=16..31) + WMMA
        #pragma unroll
        for (int mt = 0; mt < 2; ++mt) {
            const half_t* ar = &As[buf][wm * 32 + mt * 16 + lh][hi * 16];
            v8h alo = *(const v8h*)(ar);
            v8h ahi = *(const v8h*)(ar + 8);
            v16h af;
            #pragma unroll
            for (int j = 0; j < 8; ++j) { af[j] = alo[j]; af[j + 8] = ahi[j]; }
            #pragma unroll
            for (int nt = 0; nt < 4; ++nt)
                acc[mt][nt] = __builtin_amdgcn_wmma_f32_16x16x32_f16(
                    false, af, false, bfv[nt], (short)0, acc[mt][nt], false, false);
        }

        wait_async0();       // my async fills of buf^1 are in LDS
        __syncthreads();     // everyone's fills done; everyone's reads of buf done
        buf ^= 1;
    }

    // ---- epilogue: C/D 16x16 f32 layout: VGPR r -> M = hi*8+r, N = lh ----
    const int row0 = rowBase + wm * 32 + hi * 8;
    const int col0 = colBase + wn * 64 + lh;
    #pragma unroll
    for (int mt = 0; mt < 2; ++mt)
        #pragma unroll
        for (int nt = 0; nt < 4; ++nt)
            #pragma unroll
            for (int r = 0; r < 8; ++r) {
                const int row = row0 + mt * 16 + r;
                const int col = col0 + nt * 16;
                float v = acc[mt][nt][r];
                if (EPI == 0) {
                    Ch[(long)row * N + col] = (half_t)v;
                } else if (EPI == 1) {
                    v *= scale;
                    if (col > row) v = -1e30f;        // causal mask
                    Cf[(long)row * N + col] = v;
                } else if (EPI == 2) {
                    const long idx = (long)row * N + col;
                    Cf[idx] = v + extra[idx];
                } else if (EPI == 3) {
                    Ch[(long)row * N + col] = (half_t)fmaxf(v + bias[col], 0.f);
                } else if (EPI == 4) {
                    const long idx = (long)row * N + col;
                    Cf[idx] = v + bias[col] + extra[idx];
                } else {
                    Ch[(long)col * M + row] = (half_t)v;   // transposed (V^T)
                }
            }
}

// ---------------------------------------------------------------------------
// Host-side orchestration
// ---------------------------------------------------------------------------
extern "C" void kernel_launch(void* const* d_in, const int* in_sizes, int n_in,
                              void* d_out, int out_size, void* d_ws, size_t ws_size,
                              hipStream_t stream)
{
    (void)in_sizes; (void)n_in; (void)out_size; (void)ws_size;

    const float* x      = (const float*)d_in[0];
    const float* gamma1 = (const float*)d_in[1];
    const float* beta1  = (const float*)d_in[2];
    const float* w_q    = (const float*)d_in[3];
    const float* w_k    = (const float*)d_in[4];
    const float* w_v    = (const float*)d_in[5];
    const float* gamma2 = (const float*)d_in[6];
    const float* beta2  = (const float*)d_in[7];
    const float* W1     = (const float*)d_in[8];
    const float* b1     = (const float*)d_in[9];
    const float* W2     = (const float*)d_in[10];
    const float* b2     = (const float*)d_in[11];
    float* out          = (float*)d_out;

    const long BTD = (long)Bx * Tx * Dx;   // 8388608
    const long DD  = (long)Dx * Dx;        // 1048576
    const long BTT = (long)Bx * Tx * Tx;   // 16777216

    // workspace carve-up
    char* p = (char*)d_ws;
    half_t* wqT   = (half_t*)p; p += DD * 2;    // weights, transposed f16 [N][K]
    half_t* wkT   = (half_t*)p; p += DD * 2;
    half_t* wvT   = (half_t*)p; p += DD * 2;
    half_t* w1T   = (half_t*)p; p += DD * 2;
    half_t* w2T   = (half_t*)p; p += DD * 2;
    half_t* h16   = (half_t*)p; p += BTD * 2;
    half_t* q16   = (half_t*)p; p += BTD * 2;
    half_t* k16   = (half_t*)p; p += BTD * 2;
    half_t* vT    = (half_t*)p; p += BTD * 2;   // [D][B*T] (batch-interleaved)
    float*  score = (float*)p;  p += BTT * 4;
    half_t* attn  = (half_t*)p; p += BTT * 2;
    float*  y     = (float*)p;  p += BTD * 4;
    half_t* h2    = (half_t*)p; p += BTD * 2;
    half_t* a1    = (half_t*)p; p += BTD * 2;

    // 1. weights fp32 [K][N] -> f16 [N][K]
    dim3 gT(Dx / 32, Dx / 32);
    transpose_f32_to_f16<<<gT, 256, 0, stream>>>(w_q, wqT, Dx, Dx);
    transpose_f32_to_f16<<<gT, 256, 0, stream>>>(w_k, wkT, Dx, Dx);
    transpose_f32_to_f16<<<gT, 256, 0, stream>>>(w_v, wvT, Dx, Dx);
    transpose_f32_to_f16<<<gT, 256, 0, stream>>>(W1,  w1T, Dx, Dx);
    transpose_f32_to_f16<<<gT, 256, 0, stream>>>(W2,  w2T, Dx, Dx);

    // 2. LN1: x -> h16
    ln_f16_kernel<<<Bx * Tx, 256, 0, stream>>>(x, gamma1, beta1, h16, Dx);

    // 3. QKV projections: [8192,1024] x [1024,1024]^T
    dim3 gProj((Bx * Tx) / BM, Dx / BN, 1);
    gemm_wmma<0><<<gProj, 256, 0, stream>>>(
        h16, wqT, nullptr, q16, nullptr, nullptr, 1.f,
        Bx * Tx, Dx, Dx, Dx, Dx, 0, 0, 0);
    gemm_wmma<0><<<gProj, 256, 0, stream>>>(
        h16, wkT, nullptr, k16, nullptr, nullptr, 1.f,
        Bx * Tx, Dx, Dx, Dx, Dx, 0, 0, 0);
    gemm_wmma<5><<<gProj, 256, 0, stream>>>(      // V stored transposed
        h16, wvT, nullptr, vT, nullptr, nullptr, 1.f,
        Bx * Tx, Dx, Dx, Dx, Dx, 0, 0, 0);

    // 4. scores = (Q K^T) / sqrt(D) + causal mask  (batched NT)
    dim3 gScore(Tx / BM, Tx / BN, Bx);
    gemm_wmma<1><<<gScore, 256, 0, stream>>>(
        q16, k16, score, nullptr, nullptr, nullptr, 0.03125f,
        Tx, Tx, Dx, Dx, Dx, (long)Tx * Dx, (long)Tx * Dx, (long)Tx * Tx);

    // 5. softmax rows -> f16 probabilities
    softmax_f16_kernel<<<Bx * Tx, 256, 0, stream>>>(score, attn, Tx);

    // 6. y = attn @ V + x : B = V^T [D][B*T], ldb = B*T, batch offset Tx
    dim3 gPV(Tx / BM, Dx / BN, Bx);
    gemm_wmma<2><<<gPV, 256, 0, stream>>>(
        attn, vT, y, nullptr, nullptr, x, 1.f,
        Tx, Dx, Tx, Tx, Bx * Tx, (long)Tx * Tx, (long)Tx, (long)Tx * Dx);

    // 7. LN2: y -> h2
    ln_f16_kernel<<<Bx * Tx, 256, 0, stream>>>(y, gamma2, beta2, h2, Dx);

    // 8. MLP1: relu(h2 @ W1 + b1) -> f16
    gemm_wmma<3><<<gProj, 256, 0, stream>>>(
        h2, w1T, nullptr, a1, b1, nullptr, 1.f,
        Bx * Tx, Dx, Dx, Dx, Dx, 0, 0, 0);

    // 9. MLP2 + residual: out = a1 @ W2 + b2 + y  (f32 final)
    gemm_wmma<4><<<gProj, 256, 0, stream>>>(
        a1, w2T, out, nullptr, b2, y, 1.f,
        Bx * Tx, Dx, Dx, Dx, Dx, 0, 0, 0);
}
